// RNN_45114336477963
// MI455X (gfx1250) — compile-verified, matrix operand order checked by reading, not run on hardware
//
#include <hip/hip_runtime.h>
#include <hip/hip_bf16.h>
#include <math.h>

// ---------------- problem constants ----------------
constexpr int kBS  = 32;
constexpr int kSEQ = 512;
constexpr int kIN  = 1024;
constexpr int kH   = 1024;
constexpr int kL   = 2;
constexpr int kStepBlocks = 64;     // one block per 16-wide n-tile

typedef __attribute__((ext_vector_type(16))) __bf16 v16bf;
typedef __attribute__((ext_vector_type(8)))  __bf16 v8bf;
typedef __attribute__((ext_vector_type(8)))  float  v8f;

// ---------------------------------------------------------------------------
// Load one 16x32 bf16 WMMA fragment for this lane.
// `rowbase` points at element k=0 of this lane's row (A: row M=lane&15 of the
// tile; B: column N=lane&15, stored K-contiguous). Per ISA 7.12.2 the lane's
// 16 half-elements are K = {0..7, 16..23} (lanes 0-15) or {8..15, 24..31}
// (lanes 16-31): two contiguous 16-byte runs -> two global_load_b128.
// ---------------------------------------------------------------------------
__device__ __forceinline__ v16bf load_frag16x32(const __bf16* rowbase, int lane) {
    const int off0 = (lane < 16) ? 0 : 8;
    union { v16bf v; v8bf h[2]; } u;
    u.h[0] = *(const v8bf*)(rowbase + off0);
    u.h[1] = *(const v8bf*)(rowbase + off0 + 16);
    return u.v;
}

__device__ __forceinline__ v8f wmma_split(v8f acc, v16bf ah, v16bf al, v16bf bh, v16bf bl) {
    // (ah+al)*(bh+bl) ~= ah*bh + ah*bl + al*bh   (drop lo*lo term)
    acc = __builtin_amdgcn_wmma_f32_16x16x32_bf16(false, ah, false, bh, (short)0, acc, false, false);
    acc = __builtin_amdgcn_wmma_f32_16x16x32_bf16(false, ah, false, bl, (short)0, acc, false, false);
    acc = __builtin_amdgcn_wmma_f32_16x16x32_bf16(false, al, false, bh, (short)0, acc, false, false);
    return acc;
}

// ---------------- helper kernels ----------------

__global__ void k_split_f32(const float* __restrict__ src,
                            __bf16* __restrict__ hi, __bf16* __restrict__ lo, int n) {
    int i = blockIdx.x * blockDim.x + threadIdx.x;
    if (i >= n) return;
    float x = src[i];
    __bf16 h = (__bf16)x;
    hi[i] = h;
    lo[i] = (__bf16)(x - (float)h);
}

__global__ void k_bias_sum(const float* __restrict__ bih, const float* __restrict__ bhh,
                           float* __restrict__ bsum, int n) {
    int i = blockIdx.x * blockDim.x + threadIdx.x;
    if (i < n) bsum[i] = bih[i] + bhh[i];
}

// x[b,s,i] (batch-first) -> act rows r = s*BS+b, split into bf16 hi/lo
__global__ void k_reorder_split_x(const float* __restrict__ x,
                                  __bf16* __restrict__ hi, __bf16* __restrict__ lo) {
    int i = blockIdx.x * blockDim.x + threadIdx.x;      // i = r*kIN + c
    int c = i & (kIN - 1);
    int r = i >> 10;                                    // kIN = 1024
    int b = r & (kBS - 1);
    int s = r >> 5;                                     // kBS = 32
    float v = x[((size_t)b * kSEQ + s) * kIN + c];
    __bf16 h = (__bf16)v;
    hi[i] = h;
    lo[i] = (__bf16)(v - (float)h);
}

// ---------------- xproj GEMM: Y[r,n] = sum_k A[r,k]*W[n,k] + bias[n] ----------------
// 2x2 register blocking: each wave computes a 32x32 patch (4 accumulator tiles),
// 8 fragment loads -> 12 WMMAs per K-chunk. Block = 8 waves covering 64(M)x128(N).
__global__ __launch_bounds__(256)
void k_xproj_gemm(const __bf16* __restrict__ Ahi, const __bf16* __restrict__ Alo,
                  const __bf16* __restrict__ Whi, const __bf16* __restrict__ Wlo,
                  const float* __restrict__ bias, float* __restrict__ Y,
                  int N, int K) {
    const int lane = threadIdx.x & 31;
    const int wave = threadIdx.x >> 5;
    const int wm = wave & 1;
    const int wn = wave >> 1;
    const int row0 = blockIdx.y * 64 + wm * 32;
    const int col0 = blockIdx.x * 128 + wn * 32;
    const int l15 = lane & 15;

    const __bf16 *aph[2], *apl[2], *bph[2], *bpl[2];
#pragma unroll
    for (int i = 0; i < 2; ++i) {
        aph[i] = Ahi + (size_t)(row0 + i * 16 + l15) * K;
        apl[i] = Alo + (size_t)(row0 + i * 16 + l15) * K;
        bph[i] = Whi + (size_t)(col0 + i * 16 + l15) * K;
        bpl[i] = Wlo + (size_t)(col0 + i * 16 + l15) * K;
    }

    v8f acc[2][2] = {};
    for (int k = 0; k < K; k += 32) {
        v16bf ah[2], al[2], bh[2], bl[2];
#pragma unroll
        for (int i = 0; i < 2; ++i) {
            ah[i] = load_frag16x32(aph[i] + k, lane);
            al[i] = load_frag16x32(apl[i] + k, lane);
            bh[i] = load_frag16x32(bph[i] + k, lane);
            bl[i] = load_frag16x32(bpl[i] + k, lane);
        }
#pragma unroll
        for (int mi = 0; mi < 2; ++mi)
#pragma unroll
            for (int ni = 0; ni < 2; ++ni)
                acc[mi][ni] = wmma_split(acc[mi][ni], ah[mi], al[mi], bh[ni], bl[ni]);
    }

#pragma unroll
    for (int mi = 0; mi < 2; ++mi) {
        const int mbase = row0 + mi * 16 + ((lane < 16) ? 0 : 8);
#pragma unroll
        for (int ni = 0; ni < 2; ++ni) {
            const int bn = col0 + ni * 16 + l15;
            const float bv = bias[bn];
#pragma unroll
            for (int r = 0; r < 8; ++r)
                Y[(size_t)(mbase + r) * N + bn] = acc[mi][ni][r] + bv;
        }
    }
}

// ---------------- persistent recurrence kernel: one launch per layer ----------------
// h_new[b,n] = tanh(xproj[s,b,n] + sum_k h_prev[b,k]*W_hh[n,k]) for s = 0..511.
// Grid = 64 blocks (one per 16-col n-tile; co-resident: 128 thr / 8KB LDS each).
// Per step: each block computes BOTH batch m-tiles (M=32) for its n-tile, W
// fragments loaded once per 6 WMMAs; K split 4 ways across the block's waves
// (serial WMMA chain 4x shorter), partials reduced through LDS. Steps separated
// by a device-scope grid barrier (threadfence + release-arrive + per-wave
// acquire spin), eliminating 512 dependent kernel launches.
__global__ __launch_bounds__(128)
void k_rnn_layer(const float*  __restrict__ xproj,               // [SEQ][BS][H]
                 const __bf16* __restrict__ Whi, const __bf16* __restrict__ Wlo,
                 __bf16* __restrict__ h0_hi, __bf16* __restrict__ h0_lo,  // ping (zeroed: h(0)=0)
                 __bf16* __restrict__ h1_hi, __bf16* __restrict__ h1_lo,  // pong
                 float*  __restrict__ f32dst, long long dstStrideB,       // nullable (last layer out)
                 __bf16* __restrict__ act_hi, __bf16* __restrict__ act_lo,// nullable (-> next layer)
                 float*  __restrict__ hN,                                  // h_n slot for this layer
                 int*    __restrict__ counter) {                           // zeroed arrival counter
    const int K  = kH;          // 1024
    const int KS = K / 4;       // 256 per wave
    const int lane = threadIdx.x & 31;
    const int wave = threadIdx.x >> 5;
    const int nt   = blockIdx.x;            // n-tile
    const int l15  = lane & 15;
    const int k0   = wave * KS;

    const __bf16* bph = Whi + (size_t)(nt * 16 + l15) * K + k0;
    const __bf16* bpl = Wlo + (size_t)(nt * 16 + l15) * K + k0;

    __shared__ float red[4][2][32][8];      // [wave][mtile][lane][vgpr]

    for (int s = 0; s < kSEQ; ++s) {
        const __bf16* hp_hi = (s & 1) ? h1_hi : h0_hi;
        const __bf16* hp_lo = (s & 1) ? h1_lo : h0_lo;
        __bf16*       hn_hi = (s & 1) ? h0_hi : h1_hi;
        __bf16*       hn_lo = (s & 1) ? h0_lo : h1_lo;

        const __bf16* aph0 = hp_hi + (size_t)(l15)      * K + k0;   // batch rows 0..15
        const __bf16* apl0 = hp_lo + (size_t)(l15)      * K + k0;
        const __bf16* aph1 = hp_hi + (size_t)(16 + l15) * K + k0;   // batch rows 16..31
        const __bf16* apl1 = hp_lo + (size_t)(16 + l15) * K + k0;

        v8f acc0 = {}, acc1 = {};
        for (int k = 0; k < KS; k += 32) {
            v16bf bh  = load_frag16x32(bph  + k, lane);
            v16bf bl  = load_frag16x32(bpl  + k, lane);
            v16bf ah0 = load_frag16x32(aph0 + k, lane);
            v16bf al0 = load_frag16x32(apl0 + k, lane);
            v16bf ah1 = load_frag16x32(aph1 + k, lane);
            v16bf al1 = load_frag16x32(apl1 + k, lane);
            acc0 = wmma_split(acc0, ah0, al0, bh, bl);
            acc1 = wmma_split(acc1, ah1, al1, bh, bl);
        }

#pragma unroll
        for (int r = 0; r < 8; ++r) {
            red[wave][0][lane][r] = acc0[r];
            red[wave][1][lane][r] = acc1[r];
        }
        __syncthreads();

        if (wave < 2) {                     // wave w finalizes m-tile w
            float sum[8];
#pragma unroll
            for (int r = 0; r < 8; ++r)
                sum[r] = red[0][wave][lane][r] + red[1][wave][lane][r] +
                         red[2][wave][lane][r] + red[3][wave][lane][r];

            const int bn    = nt * 16 + l15;
            const int mbase = wave * 16 + ((lane < 16) ? 0 : 8);
            const float* xp = xproj + (size_t)s * kBS * kH;
#pragma unroll
            for (int r = 0; r < 8; ++r) {
                const int m = mbase + r;                        // batch index (0..31)
                const size_t hn_idx = (size_t)m * kH + bn;
                float v = tanhf(sum[r] + xp[hn_idx]);
                __bf16 vh = (__bf16)v;
                __bf16 vl = (__bf16)(v - (float)vh);
                hn_hi[hn_idx] = vh;
                hn_lo[hn_idx] = vl;
                if (f32dst) f32dst[(size_t)m * dstStrideB + (size_t)s * kH + bn] = v;
                if (act_hi) {
                    const size_t aidx = ((size_t)(s * kBS + m)) * kH + bn;
                    act_hi[aidx] = vh;
                    act_lo[aidx] = vl;
                }
                if (s == kSEQ - 1) hN[hn_idx] = v;
            }
        }

        // ---- grid-wide barrier: h(s+1) must be globally visible before s+1 ----
        __threadfence();                    // every thread: device-scope release of its stores
        __syncthreads();
        if (threadIdx.x == 0)
            __hip_atomic_fetch_add(counter, 1, __ATOMIC_RELEASE, __HIP_MEMORY_SCOPE_AGENT);
        const int target = kStepBlocks * (s + 1);
        if (lane == 0) {                    // per-wave acquire: invalidate this WGP's caches
            while (__hip_atomic_load(counter, __ATOMIC_ACQUIRE, __HIP_MEMORY_SCOPE_AGENT) < target)
                __builtin_amdgcn_s_sleep(1);
        }
        __syncthreads();
    }
}

// ---------------- host side ----------------
extern "C" void kernel_launch(void* const* d_in, const int* in_sizes, int n_in,
                              void* d_out, int out_size, void* d_ws, size_t ws_size,
                              hipStream_t stream) {
    const float* x    = (const float*)d_in[0];   // [BS, SEQ, IN]
    const float* wih  = (const float*)d_in[1];   // [L, H, IN]
    const float* whh  = (const float*)d_in[2];   // [L, H, H]
    const float* bih  = (const float*)d_in[3];   // [L, H]
    const float* bhh  = (const float*)d_in[4];   // [L, H]
    float* out = (float*)d_out;                  // [BS,SEQ,H] ++ [L,BS,H]

    // -------- workspace carve-out (256B aligned) --------
    char* p = (char*)d_ws;
    auto carve = [&](size_t bytes) -> void* {
        void* r = (void*)p;
        p += (bytes + 255) & ~(size_t)255;
        return r;
    };
    const size_t nWih = (size_t)kL * kH * kIN;
    const size_t nWhh = (size_t)kL * kH * kH;
    const size_t nAct = (size_t)kSEQ * kBS * kH;

    __bf16* wih_hi = (__bf16*)carve(nWih * 2);
    __bf16* wih_lo = (__bf16*)carve(nWih * 2);
    __bf16* whh_hi = (__bf16*)carve(nWhh * 2);
    __bf16* whh_lo = (__bf16*)carve(nWhh * 2);
    float*  bsum   = (float*) carve((size_t)kL * kH * 4);
    float*  xproj  = (float*) carve(nAct * 4);
    __bf16* act_hi = (__bf16*)carve(nAct * 2);
    __bf16* act_lo = (__bf16*)carve(nAct * 2);
    __bf16* hbuf   = (__bf16*)carve((size_t)4 * kBS * kH * 2); // {h0_hi,h0_lo,h1_hi,h1_lo}
    int*    counter= (int*)   carve(256);
    __bf16* h0_hi = hbuf;
    __bf16* h0_lo = hbuf + 1 * kBS * kH;
    __bf16* h1_hi = hbuf + 2 * kBS * kH;
    __bf16* h1_lo = hbuf + 3 * kBS * kH;

    // -------- one-time prep --------
    k_split_f32<<<(int)(nWih / 256), 256, 0, stream>>>(wih, wih_hi, wih_lo, (int)nWih);
    k_split_f32<<<(int)(nWhh / 256), 256, 0, stream>>>(whh, whh_hi, whh_lo, (int)nWhh);
    k_bias_sum<<<(kL * kH + 255) / 256, 256, 0, stream>>>(bih, bhh, bsum, kL * kH);
    k_reorder_split_x<<<(int)(nAct / 256), 256, 0, stream>>>(x, act_hi, act_lo);

    const size_t outMain = (size_t)kBS * kSEQ * kH;   // output tensor elements

    for (int l = 0; l < kL; ++l) {
        // xproj = act @ W_ih[l]^T + (b_ih+b_hh)   -> [SEQ*BS, H]
        dim3 ggrid(kH / 128, (kSEQ * kBS) / 64);
        k_xproj_gemm<<<ggrid, 256, 0, stream>>>(
            act_hi, act_lo,
            wih_hi + (size_t)l * kH * kIN, wih_lo + (size_t)l * kH * kIN,
            bsum + (size_t)l * kH, xproj, kH, kIN);

        // h(0) = 0 (h0 hi+lo contiguous) and barrier counter = 0
        hipMemsetAsync(h0_hi, 0, (size_t)2 * kBS * kH * 2, stream);
        hipMemsetAsync(counter, 0, sizeof(int), stream);

        float*  f32dst = nullptr;  long long strideB = 0;
        __bf16* ah = nullptr;      __bf16* al = nullptr;
        if (l == kL - 1) {          // last layer: fp32 output, batch-first
            f32dst = out;
            strideB = (long long)kSEQ * kH;
        } else {                    // feed next layer directly as split bf16
            ah = act_hi; al = act_lo;
        }
        float* hN = out + outMain + (size_t)l * kBS * kH;

        k_rnn_layer<<<kStepBlocks, 128, 0, stream>>>(
            xproj,
            whh_hi + (size_t)l * kH * kH, whh_lo + (size_t)l * kH * kH,
            h0_hi, h0_lo, h1_hi, h1_lo,
            f32dst, strideB, ah, al, hN, counter);
    }
}
